// ChemBFN_43937515438610
// MI455X (gfx1250) — matrix operand order repaired and checked by reading, not compiled, not run on hardware
//
#include <hip/hip_runtime.h>
#include <hip/hip_bf16.h>

typedef __attribute__((ext_vector_type(16))) _Float16 v16h;
typedef __attribute__((ext_vector_type(8)))  float    v8f;
typedef __attribute__((ext_vector_type(4)))  unsigned int v4u;
typedef __attribute__((ext_vector_type(8)))  int      v8i;
typedef __attribute__((ext_vector_type(4)))  int      v4i;

#define BATCH  16
#define SEQ    1024
#define VOCAB  256
#define CHN    512
#define NHEAD  8
#define HD     64
#define NLAYER 12

__device__ __forceinline__ float selu_f(float x) {
  const float l = 1.0507009873554805f;
  const float a = 1.6732632423543772f;
  return x > 0.f ? l * x : l * a * (__expf(x) - 1.f);
}

// Issue a TDM load of a 32-row x 64-col fp32 tile (row stride = N elements)
// from global memory into LDS at byte offset ldsOff. D# built per CDNA5 ISA
// 8.3/8.4: group0 = {ctrl(count=1), lds_addr, global_addr[56:0], type=2},
// group1 = {data_size=4B, tensor_dim0=N, tensor_dim1=K, tile 64x32, stride=N}.
// This toolchain exposes the 6-arg builtin (extra zero int32x8 group).
__device__ __forceinline__ void tdm_load_tile(const float* gsrc, unsigned ldsOff,
                                              int N, int K) {
  unsigned long long ga = (unsigned long long)(uintptr_t)gsrc;
  v4u g0;
  g0.x = 1u;                                             // count=1, user mode
  g0.y = ldsOff;                                         // lds_addr
  g0.z = (unsigned)(ga & 0xFFFFFFFFu);                   // global_addr[31:0]
  g0.w = (unsigned)((ga >> 32) & 0x01FFFFFFu) | (2u << 30);  // addr[56:32] | type=2
  v8i g1;
  g1[0] = (int)(2u << 16);                               // data_size = 2 (4 bytes)
  g1[1] = (int)(((unsigned)N & 0xFFFFu) << 16);          // tensor_dim0 lo16
  g1[2] = (int)((((unsigned)N) >> 16) | (((unsigned)K & 0xFFFFu) << 16)); // dim0 hi | dim1 lo
  g1[3] = (int)((((unsigned)K) >> 16) | (64u << 16));    // dim1 hi | tile_dim0 = 64
  g1[4] = 32;                                            // tile_dim1 = 32, tile_dim2 = 0
  g1[5] = N;                                             // tensor_dim0_stride lo32
  g1[6] = 0;
  g1[7] = 0;
  v4i gz  = {0, 0, 0, 0};
  v8i gz8 = {0, 0, 0, 0, 0, 0, 0, 0};
  __builtin_amdgcn_tensor_load_to_lds(g0, g1, gz, gz, gz8, 0);
}

// ---------------------------------------------------------------------------
// WMMA GEMM: C[M,N] = A[M,K] x B[K,N] + bias, optional SELU epilogue.
// Workgroup = 4 waves -> 64x64 output block (wave = 16x64 strip, 4 acc tiles,
// one shared A fragment per K-step). B tiles (32x64 fp32) are streamed into
// LDS by the Tensor Data Mover, double buffered (TENSORcnt pipelining).
// Fragment layouts per CDNA5 ISA 7.12.2.
// ---------------------------------------------------------------------------
template <int ACT>
__global__ __launch_bounds__(128) void gemm_wmma(
    const float* __restrict__ A, const float* __restrict__ Bm,
    const float* __restrict__ bias, float* __restrict__ C,
    int M, int N, int K)
{
  __shared__ float Bsh[2][32][64];   // 16 KB, double buffered
  int wave = threadIdx.x >> 5;
  int lane = threadIdx.x & 31;
  int blocksN = N >> 6;
  int bm = blockIdx.x / blocksN;
  int bn = blockIdx.x - bm * blocksN;
  int row0 = bm * 64 + wave * 16;
  int col0 = bn * 64;

  int n  = lane & 15;
  int kg = lane >> 4;

  const float* Arow = A + (size_t)(row0 + n) * K;
  v8f acc[4] = {};

  const unsigned ldsOff0 = (unsigned)(uintptr_t)&Bsh[0][0][0];
  const unsigned ldsOff1 = (unsigned)(uintptr_t)&Bsh[1][0][0];

  if (wave == 0)
    tdm_load_tile(Bm + (size_t)0 * N + col0, ldsOff0, N, K);

  int buf = 0;
  for (int k0 = 0; k0 < K; k0 += 32, buf ^= 1) {
    if (wave == 0) {
      if (k0 + 32 < K) {
        tdm_load_tile(Bm + (size_t)(k0 + 32) * N + col0,
                      buf ? ldsOff0 : ldsOff1, N, K);
        __builtin_amdgcn_s_wait_tensorcnt(1);  // current tile landed
      } else {
        __builtin_amdgcn_s_wait_tensorcnt(0);
      }
    }
    __syncthreads();

    // A fragment (shared by all 4 column tiles)
    v16h af;
    const float* ap = Arow + k0 + kg * 8;
#pragma unroll
    for (int i = 0; i < 8; ++i) {
      af[i]     = (_Float16)ap[i];
      af[8 + i] = (_Float16)ap[16 + i];
    }

#pragma unroll
    for (int t = 0; t < 4; ++t) {
      v16h bf;
#pragma unroll
      for (int i = 0; i < 16; ++i)
        bf[i] = (_Float16)Bsh[buf][kg * 8 + (i & 7) + (i >> 3) * 16][t * 16 + n];
      acc[t] = __builtin_amdgcn_wmma_f32_16x16x32_f16(false, af, false, bf,
                                                      (short)0, acc[t], false, false);
    }
    __syncthreads();  // block done reading this buffer before it is refilled
  }

  int mbase = kg * 8;
#pragma unroll
  for (int t = 0; t < 4; ++t) {
    float bv = bias ? bias[col0 + t * 16 + n] : 0.f;
#pragma unroll
    for (int r = 0; r < 8; ++r) {
      float v = acc[t][r] + bv;
      if (ACT == 1) v = selu_f(v);
      C[(size_t)(row0 + mbase + r) * N + col0 + t * 16 + n] = v;
    }
  }
}

// ---------------------------------------------------------------------------
// Flash attention: one wave handles 16 queries for one (batch, head).
// q,k,v in (B, NH, T, D). Online softmax over 32-key chunks.
// Scores via 2x(2 WMMA), P re-laid out through LDS, P*V via 4 WMMA.
// ---------------------------------------------------------------------------
__global__ __launch_bounds__(32) void attn_flash(
    const float* __restrict__ q, const float* __restrict__ k,
    const float* __restrict__ v, float* __restrict__ o)
{
  __shared__ float Pl[16][32];
  int lane = threadIdx.x & 31;
  int qb   = blockIdx.x * 16;
  int head = blockIdx.y;
  int b    = blockIdx.z;

  const size_t base = (size_t)(b * NHEAD + head) * SEQ * HD;
  const float* qp = q + base;
  const float* kp = k + base;
  const float* vp = v + base;

  int n  = lane & 15;
  int kg = lane >> 4;

  v16h qf[2];
#pragma unroll
  for (int c = 0; c < 2; ++c) {
    const float* qq = qp + (size_t)(qb + n) * HD + c * 32 + kg * 8;
#pragma unroll
    for (int i = 0; i < 8; ++i) {
      qf[c][i]     = (_Float16)qq[i];
      qf[c][8 + i] = (_Float16)qq[16 + i];
    }
  }

  v8f oacc[4] = {};
  float mrow[8], lrow[8];
#pragma unroll
  for (int r = 0; r < 8; ++r) { mrow[r] = -1e30f; lrow[r] = 0.f; }

  const float invTP = 0.0883883476483184f;  // 1/sqrt(2*D)

  for (int kb = 0; kb < SEQ; kb += 32) {
    v8f s[2];
#pragma unroll
    for (int st = 0; st < 2; ++st) {
      v8f z = {};
#pragma unroll
      for (int c = 0; c < 2; ++c) {
        v16h kf;
        const float* kk = kp + (size_t)(kb + st * 16 + n) * HD + c * 32 + kg * 8;
#pragma unroll
        for (int i = 0; i < 16; ++i) kf[i] = (_Float16)kk[(i & 7) + (i >> 3) * 16];
        z = __builtin_amdgcn_wmma_f32_16x16x32_f16(false, qf[c], false, kf,
                                                   (short)0, z, false, false);
      }
      s[st] = z;
    }

#pragma unroll
    for (int r = 0; r < 8; ++r) {
      float s0 = s[0][r] * invTP;
      float s1 = s[1][r] * invTP;
      float cm = fmaxf(s0, s1);
#pragma unroll
      for (int d = 1; d < 16; d <<= 1) cm = fmaxf(cm, __shfl_xor(cm, d, 16));
      float nm    = fmaxf(mrow[r], cm);
      float alpha = __expf(mrow[r] - nm);
      float p0 = __expf(s0 - nm);
      float p1 = __expf(s1 - nm);
      float rs = p0 + p1;
#pragma unroll
      for (int d = 1; d < 16; d <<= 1) rs += __shfl_xor(rs, d, 16);
      lrow[r] = lrow[r] * alpha + rs;
      mrow[r] = nm;
#pragma unroll
      for (int t = 0; t < 4; ++t) oacc[t][r] *= alpha;
      Pl[kg * 8 + r][n]      = p0;
      Pl[kg * 8 + r][16 + n] = p1;
    }
    __syncthreads();

    v16h pf;
#pragma unroll
    for (int i = 0; i < 8; ++i) {
      pf[i]     = (_Float16)Pl[n][kg * 8 + i];
      pf[8 + i] = (_Float16)Pl[n][kg * 8 + 16 + i];
    }
    __syncthreads();

#pragma unroll
    for (int t = 0; t < 4; ++t) {
      v16h vf;
      const float* vv = vp + (size_t)(kb + kg * 8) * HD + t * 16 + n;
#pragma unroll
      for (int i = 0; i < 16; ++i) vf[i] = (_Float16)vv[(size_t)((i & 7) + (i >> 3) * 16) * HD];
      oacc[t] = __builtin_amdgcn_wmma_f32_16x16x32_f16(false, pf, false, vf,
                                                       (short)0, oacc[t], false, false);
    }
  }

#pragma unroll
  for (int r = 0; r < 8; ++r) {
    float inv = 1.f / lrow[r];
    int row = qb + kg * 8 + r;
#pragma unroll
    for (int t = 0; t < 4; ++t)
      o[(size_t)(b * SEQ + row) * CHN + head * HD + t * 16 + n] = oacc[t][r] * inv;
  }
}

// --------------------------- elementwise kernels ---------------------------

__global__ __launch_bounds__(256) void ln_mod_kernel(
    const float* __restrict__ h, const float* __restrict__ mod,
    int mstride, int sh_off, int sc_off, float* __restrict__ hn)
{
  __shared__ float rs[256], rs2[256];
  int token = blockIdx.x;
  int b = token / SEQ;
  int tid = threadIdx.x;
  const float* hp = h + (size_t)token * CHN;
  float s = 0.f, ss = 0.f;
  for (int c = tid; c < CHN; c += 256) { float x = hp[c]; s += x; ss += x * x; }
  rs[tid] = s; rs2[tid] = ss;
  __syncthreads();
  for (int w = 128; w > 0; w >>= 1) {
    if (tid < w) { rs[tid] += rs[tid + w]; rs2[tid] += rs2[tid + w]; }
    __syncthreads();
  }
  float mean = rs[0] / CHN;
  float var  = rs2[0] / CHN - mean * mean;
  float rstd = rsqrtf(var + 1e-6f);
  const float* mb = mod + (size_t)b * mstride;
  for (int c = tid; c < CHN; c += 256) {
    float x = (hp[c] - mean) * rstd;
    hn[(size_t)token * CHN + c] = x * (1.f + mb[sc_off + c]) + mb[sh_off + c];
  }
}

__global__ void residual_kernel(float* __restrict__ h, const float* __restrict__ src,
                                const float* __restrict__ mod, int mstride, int goff)
{
  size_t i = (size_t)blockIdx.x * blockDim.x + threadIdx.x;
  size_t total = (size_t)BATCH * SEQ * CHN;
  if (i >= total) return;
  int c = (int)(i % CHN);
  int b = (int)(i / ((size_t)SEQ * CHN));
  h[i] += mod[(size_t)b * mstride + goff + c] * src[i];
}

__global__ void rope_kernel(const float* __restrict__ qkv,
                            float* __restrict__ qr, float* __restrict__ kr,
                            float* __restrict__ vr)
{
  size_t i = (size_t)blockIdx.x * blockDim.x + threadIdx.x;
  size_t total = (size_t)BATCH * SEQ * CHN;
  if (i >= total) return;
  int c  = (int)(i % CHN);
  int t  = (int)((i / CHN) % SEQ);
  int b  = (int)(i / ((size_t)SEQ * CHN));
  int hh = c >> 6;
  int d  = c & 63;
  const float* base = qkv + (size_t)(b * SEQ + t) * (3 * CHN);
  float qv = base[c], kv = base[CHN + c], vv = base[2 * CHN + c];
  float qpv = base[c ^ 1], kpv = base[CHN + (c ^ 1)];
  float sign = (d & 1) ? 1.f : -1.f;
  float iv = (float)(d & ~1) / (float)HD;
  float theta = __powf(10000.f, -iv);
  float pos = (float)t;
  float ang = pos * theta;
  float cs = __cosf(ang), sn = __sinf(ang);
  float zeta = __powf((iv + 0.4f) / 1.4f, pos / (float)CHN);
  size_t oidx = ((size_t)(b * NHEAD + hh) * SEQ + t) * HD + d;
  qr[oidx] = (qv * cs + sign * qpv * sn) * zeta;
  kr[oidx] = (kv * cs + sign * kpv * sn) / zeta;
  vr[oidx] = vv;
}

__global__ void small_gemm_kernel(const float* __restrict__ A, const float* __restrict__ W,
                                  const float* __restrict__ bias, float* __restrict__ out,
                                  int K, int N)
{
  int b = blockIdx.y;
  int j = blockIdx.x * blockDim.x + threadIdx.x;
  if (j >= N) return;
  float s = bias[j];
  const float* a = A + (size_t)b * K;
  for (int kk = 0; kk < K; ++kk) s += a[kk] * W[(size_t)kk * N + j];
  out[(size_t)b * N + j] = s;
}

__global__ void time_embed_kernel(const float* __restrict__ t,
                                  const float* __restrict__ W1, const float* __restrict__ b1,
                                  const float* __restrict__ W2, const float* __restrict__ b2,
                                  float* __restrict__ c)
{
  int b = blockIdx.x;
  int j = threadIdx.x;  // 512 threads
  float tv = t[b];
  float s = b2[j];
  for (int kk = 0; kk < CHN / 2; ++kk) {
    float h1 = selu_f(tv * W1[kk] + b1[kk]);
    s += h1 * W2[(size_t)kk * CHN + j];
  }
  c[(size_t)b * CHN + j] = s;
}

__global__ void selu_map_kernel(const float* __restrict__ in, float* __restrict__ out, int nelem)
{
  int i = blockIdx.x * blockDim.x + threadIdx.x;
  if (i < nelem) out[i] = selu_f(in[i]);
}

// ---------------------------------------------------------------------------

extern "C" void kernel_launch(void* const* d_in, const int* in_sizes, int n_in,
                              void* d_out, int out_size, void* d_ws, size_t ws_size,
                              hipStream_t stream)
{
  (void)in_sizes; (void)n_in; (void)out_size; (void)ws_size;

  const float* x      = (const float*)d_in[0];
  const float* t      = (const float*)d_in[1];
  const float* emb_W  = (const float*)d_in[2];
  const float* emb_b  = (const float*)d_in[3];
  const float* te_W1  = (const float*)d_in[4];
  const float* te_b1  = (const float*)d_in[5];
  const float* te_W2  = (const float*)d_in[6];
  const float* te_b2  = (const float*)d_in[7];
  const float* qkv_W  = (const float*)d_in[8];
  const float* qkv_b  = (const float*)d_in[9];
  const float* ffn_W1 = (const float*)d_in[10];
  const float* ffn_b1 = (const float*)d_in[11];
  const float* ffn_W2 = (const float*)d_in[12];
  const float* ffn_b2 = (const float*)d_in[13];
  const float* ada_W  = (const float*)d_in[14];
  const float* ada_b  = (const float*)d_in[15];
  const float* fin_W  = (const float*)d_in[16];
  const float* fin_b  = (const float*)d_in[17];
  const float* fada_W = (const float*)d_in[18];
  const float* fada_b = (const float*)d_in[19];

  float* ws = (float*)d_ws;
  const size_t TOK = (size_t)BATCH * SEQ;  // 16384
  const size_t HSZ = TOK * CHN;            // 8,388,608 floats
  size_t off = 0;
  float* h    = ws + off; off += HSZ;
  float* hn   = ws + off; off += HSZ;
  float* qr   = ws + off; off += HSZ;
  float* kr   = ws + off; off += HSZ;
  float* vr   = ws + off; off += HSZ;
  float* attn = ws + off; off += HSZ;
  float* ff2  = ws + off; off += HSZ;
  float* big  = ws + off; off += TOK * 4 * CHN;  // shared: qkv (3*CHN) / ffn hidden (4*CHN)
  float* cemb = ws + off; off += (size_t)BATCH * CHN;
  float* scc  = ws + off; off += (size_t)BATCH * CHN;
  float* mod  = ws + off; off += (size_t)BATCH * 6 * CHN;
  float* fmod = ws + off; off += (size_t)BATCH * 2 * CHN;

  time_embed_kernel<<<BATCH, CHN, 0, stream>>>(t, te_W1, te_b1, te_W2, te_b2, cemb);
  selu_map_kernel<<<(BATCH * CHN + 255) / 256, 256, 0, stream>>>(cemb, scc, BATCH * CHN);

  // h = x @ emb_W + emb_b
  {
    int M = (int)TOK, N = CHN, K = VOCAB;
    gemm_wmma<0><<<(M / 64) * (N / 64), 128, 0, stream>>>(x, emb_W, emb_b, h, M, N, K);
  }

  const int ew_blocks = (int)((TOK * CHN + 255) / 256);

  for (int l = 0; l < NLAYER; ++l) {
    small_gemm_kernel<<<dim3((6 * CHN) / 256, BATCH), 256, 0, stream>>>(
        scc, ada_W + (size_t)l * CHN * 6 * CHN, ada_b + (size_t)l * 6 * CHN,
        mod, CHN, 6 * CHN);

    ln_mod_kernel<<<(int)TOK, 256, 0, stream>>>(h, mod, 6 * CHN, 0, CHN, hn);

    {
      int M = (int)TOK, N = 3 * CHN, K = CHN;
      gemm_wmma<0><<<(M / 64) * (N / 64), 128, 0, stream>>>(
          hn, qkv_W + (size_t)l * CHN * 3 * CHN, qkv_b + (size_t)l * 3 * CHN, big, M, N, K);
    }
    rope_kernel<<<ew_blocks, 256, 0, stream>>>(big, qr, kr, vr);
    attn_flash<<<dim3(SEQ / 16, NHEAD, BATCH), 32, 0, stream>>>(qr, kr, vr, attn);
    residual_kernel<<<ew_blocks, 256, 0, stream>>>(h, attn, mod, 6 * CHN, 2 * CHN);  // g

    ln_mod_kernel<<<(int)TOK, 256, 0, stream>>>(h, mod, 6 * CHN, 3 * CHN, 4 * CHN, hn);

    {
      int M = (int)TOK, N = 4 * CHN, K = CHN;
      gemm_wmma<1><<<(M / 64) * (N / 64), 128, 0, stream>>>(
          hn, ffn_W1 + (size_t)l * CHN * 4 * CHN, ffn_b1 + (size_t)l * 4 * CHN, big, M, N, K);
    }
    {
      int M = (int)TOK, N = CHN, K = 4 * CHN;
      gemm_wmma<0><<<(M / 64) * (N / 64), 128, 0, stream>>>(
          big, ffn_W2 + (size_t)l * 4 * CHN * CHN, ffn_b2 + (size_t)l * CHN, ff2, M, N, K);
    }
    residual_kernel<<<ew_blocks, 256, 0, stream>>>(h, ff2, mod, 6 * CHN, 5 * CHN);  // gf
  }

  small_gemm_kernel<<<dim3((2 * CHN) / 256, BATCH), 256, 0, stream>>>(
      scc, fada_W, fada_b, fmod, CHN, 2 * CHN);
  ln_mod_kernel<<<(int)TOK, 256, 0, stream>>>(h, fmod, 2 * CHN, 0, CHN, hn);
  {
    int M = (int)TOK, N = VOCAB, K = CHN;
    gemm_wmma<0><<<(M / 64) * (N / 64), 128, 0, stream>>>(
        hn, fin_W, fin_b, (float*)d_out, M, N, K);
  }
}
